// RecurrentConvBlock_24592982737296
// MI455X (gfx1250) — compile-verified, matrix operand order checked by reading, not run on hardware
//
#include <hip/hip_runtime.h>
#include <math.h>

// ---------------------------------------------------------------------------
// ConvLSTM2D x3 for MI455X (gfx1250): convs as implicit-GEMM on bf16 WMMA.
// z = conv3x3(x, Wx) + conv3x3(h, Wh) + b  ->  v_wmma_f32_16x16x32_bf16
// Channels padded 48->64 so K per tap is exactly two 32-wide WMMA K-steps.
// Per-WG weight slice (Wx+Wh, 64 couts x K=576, 144 KB) staged in LDS
// (CDNA5 WGP has 320 KB); each wave holds 4 accumulators (16x64 tile) so
// consecutive WMMAs are independent (no hazard NOPs) and A-fragments are
// reused 4x.
// ---------------------------------------------------------------------------

typedef __bf16 bf16;
typedef __attribute__((ext_vector_type(16))) __bf16        v16bf;
typedef __attribute__((ext_vector_type(8)))  float         v8f;
typedef __attribute__((ext_vector_type(4)))  unsigned int  v4u;

union FragU { v4u u[2]; v16bf v; };   // 32 bytes = one 16x32 bf16 fragment slice per lane

// problem dims
constexpr int  B_  = 8, T_ = 10, H_ = 64, W_ = 64, F_ = 48;
constexpr int  CP  = 64;                 // padded channels
constexpr int  G   = 4 * F_;             // 192 gate channels
constexpr int  HW  = H_ * W_;            // 4096
constexpr int  PIX = B_ * HW;            // 32768 pixels per frame
constexpr long FRAME     = (long)PIX * CP;        // bf16 elems per padded frame
constexpr long SEQ_ELEMS = (long)T_ * FRAME;      // bf16 elems per padded sequence
constexpr int  KROW = 9 * CP;            // 576: weight row length per cout

// conv tiling
constexpr int  NT    = 64;               // couts per WG (4 x 16 per wave)
constexpr int  MT    = 128;              // pixels per WG (8 waves x 16)
constexpr int  WELEM = NT * KROW;        // 36864 bf16 per weight tile
constexpr size_t LDS_BYTES = 2 * (size_t)WELEM * sizeof(bf16);  // 147456 (< 320KB WGP LDS)

__device__ __forceinline__ float sigm(float x) { return 1.0f / (1.0f + __expf(-x)); }

// ---------------------------------------------------------------------------
// Fused gate conv: Z[pix][cout] = bias[cout] + sum_tap sum_cin
//                  ( X[pix+tap][cin]*Wx[cout][tap,cin] + H[pix+tap][cin]*Wh[...] )
// ---------------------------------------------------------------------------
__global__ __launch_bounds__(256)
void conv_gates_wmma(const bf16* __restrict__ Xin,   // (B,H,W,64) bf16 frame
                     const bf16* __restrict__ Hin,   // (B,H,W,64) bf16 hidden
                     const bf16* __restrict__ Wx,    // [192][576] bf16
                     const bf16* __restrict__ Wh,    // [192][576] bf16
                     const float* __restrict__ bias, // [192]
                     float* __restrict__ Z)          // (B,H,W,192) f32
{
  extern __shared__ bf16 smem[];        // [2][NT*KROW]: Wx slice then Wh slice

  const int tid   = threadIdx.x;
  const int lane  = tid & 31;
  const int wave  = tid >> 5;
  const int l16   = lane & 15;
  const int khalf = lane >> 4;          // K-half select per ISA A/B lane layout

  const int coutBase = blockIdx.y * NT;

  // ---- cooperative weight stage: global -> LDS, b128 granularity ----
  {
    const v4u* gx = (const v4u*)(Wx + (size_t)coutBase * KROW);
    const v4u* gh = (const v4u*)(Wh + (size_t)coutBase * KROW);
    v4u* lx = (v4u*)smem;
    v4u* lh = (v4u*)(smem + WELEM);
    const int nchunk = WELEM / 8;       // 4608 x 16B per tile
    for (int i = tid; i < nchunk; i += 256) {
      lx[i] = gx[i];
      lh[i] = gh[i];
    }
  }
  __syncthreads();

  const int mbase = blockIdx.x * MT + wave * 16;
  const int pix   = mbase + l16;        // A fragment: row M = lane%16
  const int b     = pix >> 12;
  const int y     = (pix >> 6) & 63;
  const int x     = pix & 63;

  v8f acc[4];
#pragma unroll
  for (int n = 0; n < 4; ++n)
    acc[n] = (v8f)bias[coutBase + n * 16 + l16];   // splat

  const v4u* lw = (const v4u*)smem;     // LDS as 16B chunks

  for (int tap = 0; tap < 9; ++tap) {
    const int dy = tap / 3 - 1;
    const int dx = tap - (tap / 3) * 3 - 1;
    const int yy = y + dy;
    const int xx = x + dx;
    const bool valid = ((unsigned)yy < 64u) && ((unsigned)xx < 64u);
    const long poff = ((((long)b << 6) + yy) * 64 + xx) * 64;   // elem offset

#pragma unroll
    for (int c0 = 0; c0 < CP; c0 += 32) {
      FragU ax, ah;
      if (valid) {
        // lane<16: K {0..7,16..23}; lane>=16: K {8..15,24..31} of this chunk
        ax.u[0] = *(const v4u*)(Xin + poff + c0 + khalf * 8);
        ax.u[1] = *(const v4u*)(Xin + poff + c0 + 16 + khalf * 8);
        ah.u[0] = *(const v4u*)(Hin + poff + c0 + khalf * 8);
        ah.u[1] = *(const v4u*)(Hin + poff + c0 + 16 + khalf * 8);
      } else {
        ax.u[0] = (v4u)0u; ax.u[1] = (v4u)0u;
        ah.u[0] = (v4u)0u; ah.u[1] = (v4u)0u;
      }

      const int kb = (tap * CP + c0 + khalf * 8) >> 3;   // 16B index within K row

      // x-conv: 4 independent WMMAs reusing ax
#pragma unroll
      for (int n = 0; n < 4; ++n) {
        const int rowb = ((n * 16 + l16) * KROW) >> 3;
        FragU wb;
        wb.u[0] = lw[rowb + kb];
        wb.u[1] = lw[rowb + kb + 2];
        acc[n] = __builtin_amdgcn_wmma_f32_16x16x32_bf16(false, ax.v, false, wb.v,
                                                         (short)0, acc[n], false, false);
      }
      // h-conv: 4 independent WMMAs reusing ah
#pragma unroll
      for (int n = 0; n < 4; ++n) {
        const int rowb = (WELEM + (n * 16 + l16) * KROW) >> 3;
        FragU wb;
        wb.u[0] = lw[rowb + kb];
        wb.u[1] = lw[rowb + kb + 2];
        acc[n] = __builtin_amdgcn_wmma_f32_16x16x32_bf16(false, ah.v, false, wb.v,
                                                         (short)0, acc[n], false, false);
      }
    }
  }

  // C/D layout: VGPR r holds row M = r + 8*(lane/16), col N = lane%16
#pragma unroll
  for (int n = 0; n < 4; ++n) {
    const int cout = coutBase + n * 16 + l16;
#pragma unroll
    for (int r = 0; r < 8; ++r) {
      const int prow = mbase + 8 * khalf + r;
      Z[(long)prow * G + cout] = acc[n][r];
    }
  }
}

// ---------------------------------------------------------------------------
// LSTM pointwise update. idx over PIX*64 (pad lanes keep the bf16 pad zeroed).
// ---------------------------------------------------------------------------
__global__ __launch_bounds__(256)
void lstm_pointwise(const float* __restrict__ Z, float* __restrict__ Cst,
                    bf16* __restrict__ Hprev, bf16* __restrict__ SeqOut,
                    int reluSeq, const bf16* __restrict__ Res,
                    float* __restrict__ Fout, int t)
{
  const int idx = blockIdx.x * 256 + threadIdx.x;
  if (idx >= PIX * CP) return;
  const int f   = idx & 63;
  const int pix = idx >> 6;
  if (f >= F_) {                       // keep channel pad zero for next conv
    Hprev[idx] = (bf16)0.0f;
    if (SeqOut) SeqOut[idx] = (bf16)0.0f;
    return;
  }
  const long zb = (long)pix * G;
  const float zi = Z[zb + f];
  const float zf = Z[zb + F_ + f];
  const float zg = Z[zb + 2 * F_ + f];
  const float zo = Z[zb + 3 * F_ + f];
  const long ci = (long)pix * F_ + f;
  const float c  = Cst[ci];
  const float cn = sigm(zf) * c + sigm(zi) * tanhf(zg);
  const float hv = sigm(zo) * tanhf(cn);
  Cst[ci]    = cn;
  Hprev[idx] = (bf16)hv;
  if (SeqOut) SeqOut[idx] = (bf16)(reluSeq ? fmaxf(hv, 0.0f) : hv);
  if (Fout) {                          // layer 3: residual + relu -> (B,T,H,W,F)
    const float r  = hv + (float)Res[idx];
    const int   bI = pix >> 12;
    const int   yx = pix & 4095;
    Fout[(((long)bI * T_ + t) * HW + yx) * F_ + f] = fmaxf(r, 0.0f);
  }
}

// X (B,T,H,W,48) f32  ->  S0 (T,B,H,W,64) bf16, channel-padded
__global__ __launch_bounds__(256)
void cvt_x_kernel(const float* __restrict__ X, bf16* __restrict__ S0)
{
  const long idx = (long)blockIdx.x * 256 + threadIdx.x;
  if (idx >= SEQ_ELEMS) return;
  const int c  = (int)(idx & 63);
  long r = idx >> 6;
  const int yx = (int)(r & 4095); r >>= 12;
  const int b  = (int)(r % B_);
  const int t  = (int)(r / B_);
  float v = 0.0f;
  if (c < F_) v = X[(((long)b * T_ + t) * HW + yx) * F_ + c];
  S0[idx] = (bf16)v;
}

// W (3,3,48,192) f32 -> Wt [192][tap*64+cin] bf16 (transposed, channel-padded)
__global__ __launch_bounds__(256)
void cvt_w_kernel(const float* __restrict__ Wsrc, bf16* __restrict__ Wt)
{
  const int idx = blockIdx.x * 256 + threadIdx.x;
  if (idx >= G * KROW) return;
  const int cout = idx / KROW;
  const int kk   = idx % KROW;
  const int tap  = kk >> 6;
  const int cin  = kk & 63;
  float v = 0.0f;
  if (cin < F_) v = Wsrc[((long)tap * F_ + cin) * G + cout];
  Wt[idx] = (bf16)v;
}

__global__ __launch_bounds__(256)
void fill_zero_u32(unsigned int* __restrict__ p, long n)
{
  const long i = (long)blockIdx.x * 256 + threadIdx.x;
  if (i < n) p[i] = 0u;
}

// ---------------------------------------------------------------------------
extern "C" void kernel_launch(void* const* d_in, const int* in_sizes, int n_in,
                              void* d_out, int out_size, void* d_ws, size_t ws_size,
                              hipStream_t stream)
{
  (void)in_sizes; (void)n_in; (void)out_size; (void)ws_size;

  const size_t SBYTES = (size_t)SEQ_ELEMS * sizeof(bf16);   // 41.9 MB per sequence
  const size_t ZBYTES = (size_t)PIX * G  * sizeof(float);   // 25.2 MB gate buffer
  const size_t CBYTES = (size_t)PIX * F_ * sizeof(float);   // 6.3 MB cell state
  const size_t HBYTES = (size_t)PIX * CP * sizeof(bf16);    // 4.2 MB hidden state
  const size_t WBYTES = (size_t)G * KROW * sizeof(bf16);    // 216 KB per weight

  char* p = (char*)d_ws;
  bf16*  S0 = (bf16*)p;  p += SBYTES;   // bf16 input sequence (T,B,H,W,64)
  bf16*  S1 = (bf16*)p;  p += SBYTES;   // layer1 output (= Y_c residual source)
  bf16*  S2 = (bf16*)p;  p += SBYTES;   // relu(layer2 output)
  float* Z  = (float*)p; p += ZBYTES;
  float* Cs = (float*)p; p += CBYTES;
  bf16*  Hp = (bf16*)p;  p += HBYTES;
  bf16*  Wt[6];
  for (int i = 0; i < 6; ++i) { Wt[i] = (bf16*)p; p += WBYTES; }

  // stage inputs to bf16 / WMMA layouts
  cvt_x_kernel<<<(int)((SEQ_ELEMS + 255) / 256), 256, 0, stream>>>(
      (const float*)d_in[0], S0);
  {
    static const int slot[6] = {1, 2, 4, 5, 7, 8};  // Wx1,Wh1,Wx2,Wh2,Wx3,Wh3
    const int wn = G * KROW;
    for (int i = 0; i < 6; ++i)
      cvt_w_kernel<<<(wn + 255) / 256, 256, 0, stream>>>(
          (const float*)d_in[slot[i]], Wt[i]);
  }

  const bf16*  Sin[3]    = {S0, S1, S2};
  bf16*        Sout[3]   = {S1, S2, nullptr};
  const float* biases[3] = {(const float*)d_in[3], (const float*)d_in[6],
                            (const float*)d_in[9]};

  const dim3 cgrid(PIX / MT, G / NT);                 // (256, 3)
  const int  pwBlocks = (PIX * CP + 255) / 256;

  for (int layer = 0; layer < 3; ++layer) {
    fill_zero_u32<<<(int)((HBYTES / 4 + 255) / 256), 256, 0, stream>>>(
        (unsigned int*)Hp, (long)(HBYTES / 4));
    fill_zero_u32<<<(int)((CBYTES / 4 + 255) / 256), 256, 0, stream>>>(
        (unsigned int*)Cs, (long)(CBYTES / 4));
    const bf16* WtX = Wt[layer * 2];
    const bf16* WtH = Wt[layer * 2 + 1];
    for (int t = 0; t < T_; ++t) {
      const bf16* xf = Sin[layer] + (long)t * FRAME;
      conv_gates_wmma<<<cgrid, 256, LDS_BYTES, stream>>>(xf, Hp, WtX, WtH,
                                                         biases[layer], Z);
      bf16*       so  = Sout[layer] ? Sout[layer] + (long)t * FRAME : nullptr;
      const bf16* res = (layer == 2) ? (S1 + (long)t * FRAME) : nullptr;
      float*      fo  = (layer == 2) ? (float*)d_out : nullptr;
      lstm_pointwise<<<pwBlocks, 256, 0, stream>>>(Z, Cs, Hp, so,
                                                   (layer == 1) ? 1 : 0,
                                                   res, fo, t);
    }
  }
}
